// DGCNN_72541997630000
// MI455X (gfx1250) — matrix operand order repaired and checked by reading, not compiled, not run on hardware
//
#include <hip/hip_runtime.h>
#include <hip/hip_bf16.h>

// ---------------------------------------------------------------------------
// DGCNN for MI455X (gfx1250, wave32, WMMA, 320KB LDS, async-to-LDS).
// - Heavy GEMMs (GCN x@W, conv1-as-GEMM, lin1) use v_wmma_f32_16x16x32_f16
//   with register N-blocking (GCN/lin1 8-way: one wave -> 16x128 strip, A loaded once; conv1 1-way).
// - Edge aggregation is fused per-graph: stage h tile (128KB) into LDS with
//   global_load_async_to_lds_b128 (ASYNCcnt), accumulate with ds_add_f32 LDS
//   atomics, finalize (self-loop + bias + tanh) in the same kernel.
// - Sort-pool is a bitonic sort in LDS per graph.
// ---------------------------------------------------------------------------

typedef __attribute__((ext_vector_type(16))) _Float16 v16h;
typedef __attribute__((ext_vector_type(8)))  _Float16 v8h;
typedef __attribute__((ext_vector_type(8)))  float    v8f;

#define NB    512
#define NPG   256
#define NTOT  131072
#define EDG   2097152
#define DEG   16
#define HID   128
#define TLD   385
#define KTOP  60
#define PK    416          // 385 padded to multiple of 32
#define MPOOL (NB * KTOP)  // 30720

// ---------------- degree -> dinv, fused per graph (LDS atomics) ------------
__global__ void k_deg_graph(const float* __restrict__ ew,
                            const int* __restrict__ dst,
                            float* __restrict__ dinv) {
  __shared__ float sdeg[NPG];
  const int g = blockIdx.x, t = threadIdx.x;       // 256 threads
  sdeg[t] = 1.0f;                                  // self-loop weight
  __syncthreads();
  const int ebase = g * NPG * DEG;
  for (int eo = t; eo < NPG * DEG; eo += NPG)
    unsafeAtomicAdd(&sdeg[dst[ebase + eo] - g * NPG], ew[ebase + eo]);
  __syncthreads();
  float d = sdeg[t];
  dinv[g * NPG + t] = d > 0.f ? rsqrtf(fmaxf(d, 1e-12f)) : 0.f;
}

// ---------------- embedding gather ----------------
__global__ void k_embed(const float* z_table, const int* z, _Float16* xh) {
  int idx = blockIdx.x * 256 + threadIdx.x;        // NTOT*HID
  int i = idx >> 7, c = idx & 127;
  if (i < NTOT) xh[idx] = (_Float16)z_table[(size_t)z[i] * HID + c];
}

// ---------------- weight convert/transpose to f16 ----------------
// W is [K][N] row-major (x @ W); produce Wt[n][k] f16.
__global__ void k_convert_wt(const float* W, _Float16* Wt, int K, int N) {
  int idx = blockIdx.x * 256 + threadIdx.x;
  if (idx >= K * N) return;
  int n = idx / K, k = idx - n * K;
  Wt[idx] = (_Float16)W[(size_t)k * N + n];
}
// conv1_w [16][385] -> padded [16][416] f16 (already "Bt" layout).
__global__ void k_conv1_wt(const float* w1, _Float16* w1t) {
  int idx = blockIdx.x * 256 + threadIdx.x;        // 16*416
  if (idx >= 16 * PK) return;
  int o = idx / PK, t = idx - o * PK;
  w1t[idx] = (t < TLD) ? (_Float16)w1[(size_t)o * TLD + t] : (_Float16)0.f;
}

// ---------------- WMMA GEMM: C[M,N] = A[M,K](f16) x Bt[N,K](f16)^T ----------
// M % 64 == 0, N % (16*NBLK) == 0, K % 32 == 0. 4 waves/block.
// Each wave computes a 16 x (16*NBLK) strip: A fragment loaded once per K
// chunk and reused across NBLK back-to-back v_wmma ops (register blocking).
template <int NBLK>
__global__ void k_wmma_gemm(const _Float16* __restrict__ A,
                            const _Float16* __restrict__ Bt,
                            float* __restrict__ C, int M, int N, int K) {
  const int lane = threadIdx.x & 31;
  const int wid  = threadIdx.x >> 5;
  const int half = lane >> 4;       // 0: lanes 0-15, 1: lanes 16-31
  const int lidx = lane & 15;
  const long m0 = ((long)blockIdx.x * 4 + wid) * 16;
  const long n0 = (long)blockIdx.y * 16 * NBLK;
  // A fragment (16x32 f16): lane row m0+lidx; element e<8 -> K=half*8+e,
  // e>=8 -> K=half*8+8+e : two contiguous 8-half runs at +0 and +16.
  const _Float16* arow = A + (m0 + lidx) * (long)K + half * 8;
  // B fragment (32x16 f16) from Bt[n][k]: lane col n+lidx; same K mapping as A:
  // two contiguous 8-half runs at +0 and +16.
  const _Float16* brow = Bt + (n0 + lidx) * (long)K + half * 8;
  v8f acc[NBLK];
#pragma unroll
  for (int nb = 0; nb < NBLK; ++nb) acc[nb] = (v8f){};
  for (int k0 = 0; k0 < K; k0 += 32) {
    v8h alo = *(const v8h*)(arow + k0);
    v8h ahi = *(const v8h*)(arow + k0 + 16);
    v16h a = __builtin_shufflevector(alo, ahi,
                                     0,1,2,3,4,5,6,7,8,9,10,11,12,13,14,15);
#pragma unroll
    for (int nb = 0; nb < NBLK; ++nb) {
      const _Float16* bp = brow + (long)nb * 16 * K + k0;
      v8h blo = *(const v8h*)(bp);
      v8h bhi = *(const v8h*)(bp + 16);
      v16h b = __builtin_shufflevector(blo, bhi,
                                       0,1,2,3,4,5,6,7,8,9,10,11,12,13,14,15);
      acc[nb] = __builtin_amdgcn_wmma_f32_16x16x32_f16(false, a, false, b,
                                                       (short)0, acc[nb],
                                                       false, false);
    }
  }
  // D: VGPR r -> (m0 + half*8 + r, n + lidx)
  float* crow = C + (m0 + half * 8) * (long)N + n0 + lidx;
#pragma unroll
  for (int nb = 0; nb < NBLK; ++nb) {
#pragma unroll
    for (int r = 0; r < 8; ++r) crow[(long)r * N + nb * 16] = acc[nb][r];
  }
}

// ---------------- layer 3 (HID -> 1) ----------------
__global__ void k_dense_w3(const _Float16* xh, const float* W3, float* h) {
  int i = blockIdx.x * 256 + threadIdx.x;
  if (i >= NTOT) return;
  const _Float16* xr = xh + (size_t)i * HID;
  float s = 0.f;
#pragma unroll 8
  for (int c = 0; c < HID; ++c) s += (float)xr[c] * W3[c];
  h[i] = s;
}

// ---------------- fused GCN aggregation per graph (C = 128) ----------------
// LDS: hs[256*128] staged via global_load_async_to_lds_b128 (128KB) +
//      as[256*128] accumulators (128KB) = 256KB dynamic LDS (<= 320KB/WGP).
// agg = sum_edges dinv[s]*ew*dinv[d] * h[s]  +  dinv[i]^2 * h[i]
// out = tanh(agg + bias);  writes xh (next layer, f16) and cat slice (f16).
__global__ void k_gcn_agg_fused(const float* __restrict__ h,
                                const float* __restrict__ ew,
                                const int* __restrict__ src,
                                const int* __restrict__ dst,
                                const float* __restrict__ dinv,
                                const float* __restrict__ bias,
                                _Float16* __restrict__ xh_next,
                                _Float16* __restrict__ cat,
                                int off) {
  extern __shared__ float lds[];
  float* hs = lds;                 // [NPG*HID]
  float* as = lds + NPG * HID;     // [NPG*HID]
  __shared__ float sdv[NPG];
  const int g = blockIdx.x;
  const int t = threadIdx.x;       // 256 threads
  const int node0 = g * NPG;

  sdv[t] = dinv[node0 + t];
  for (int i = t; i < NPG * HID; i += 256) as[i] = 0.f;

  // async-stage h[node0 .. node0+255][0..127] -> hs (8192 x b128)
  {
    const uint32_t lbase = (uint32_t)(uintptr_t)hs;  // LDS byte offset
    const float* gsrc = h + (size_t)node0 * HID;
    for (int i = t; i < NPG * HID / 4; i += 256) {
      uint32_t laddr = lbase + (uint32_t)i * 16u;
      const float* gp = gsrc + (size_t)i * 4;
      asm volatile("global_load_async_to_lds_b128 %0, %1, off"
                   :: "v"(laddr), "v"(gp) : "memory");
    }
    asm volatile("s_wait_asynccnt 0" ::: "memory");
  }
  __syncthreads();

  // edges of this graph: 4096 edges, 32 lanes/edge, 4 channels/lane
  const int ebase = g * NPG * DEG;
  const int c0 = (t & 31) * 4;
  for (int eo = t >> 5; eo < NPG * DEG; eo += 8) {
    const int e = ebase + eo;
    const int sl = src[e] - node0;
    const int dl = dst[e] - node0;
    const float w = sdv[sl] * ew[e] * sdv[dl];
    const float4 v = *(const float4*)(hs + sl * HID + c0);
    float* ar = as + dl * HID + c0;
    unsafeAtomicAdd(ar + 0, v.x * w);
    unsafeAtomicAdd(ar + 1, v.y * w);
    unsafeAtomicAdd(ar + 2, v.z * w);
    unsafeAtomicAdd(ar + 3, v.w * w);
  }
  __syncthreads();

  // finalize: tanh(as + hs*dinv^2 + bias)
  for (int i = t; i < NPG * HID; i += 256) {
    const int j = i >> 7, c = i & 127;
    const float dv = sdv[j];
    const float v = tanhf(as[i] + hs[i] * dv * dv + bias[c]);
    const size_t node = node0 + j;
    cat[node * TLD + off + c] = (_Float16)v;
    xh_next[node * (size_t)HID + c] = (_Float16)v;
  }
}

// ---------------- fused GCN aggregation per graph (C = 1, last layer) ------
__global__ void k_gcn_agg1(const float* __restrict__ h,
                           const float* __restrict__ ew,
                           const int* __restrict__ src,
                           const int* __restrict__ dst,
                           const float* __restrict__ dinv,
                           const float* __restrict__ bias,
                           _Float16* __restrict__ cat,
                           float* __restrict__ key) {
  __shared__ float hs[NPG];
  __shared__ float as[NPG];
  __shared__ float sdv[NPG];
  const int g = blockIdx.x, t = threadIdx.x;   // 256 threads
  const int node0 = g * NPG;
  hs[t] = h[node0 + t];
  as[t] = 0.f;
  sdv[t] = dinv[node0 + t];
  __syncthreads();
  const int ebase = g * NPG * DEG;
  for (int eo = t; eo < NPG * DEG; eo += NPG) {
    const int e = ebase + eo;
    const int sl = src[e] - node0;
    const int dl = dst[e] - node0;
    unsafeAtomicAdd(&as[dl], hs[sl] * sdv[sl] * ew[e] * sdv[dl]);
  }
  __syncthreads();
  const float dv = sdv[t];
  const float v = tanhf(as[t] + hs[t] * dv * dv + bias[0]);
  cat[(size_t)(node0 + t) * TLD + (TLD - 1)] = (_Float16)v;
  key[node0 + t] = v;
}

// ---------------- sort-pool (bitonic, stable tie-break) ----------------
__global__ void k_sortpool(const float* key, const _Float16* cat,
                           _Float16* pooled) {
  __shared__ float sk[NPG];
  __shared__ int   si[NPG];
  const int g = blockIdx.x, t = threadIdx.x;
  sk[t] = key[g * NPG + t];
  si[t] = t;
  __syncthreads();
  for (int ksz = 2; ksz <= NPG; ksz <<= 1) {
    for (int j = ksz >> 1; j > 0; j >>= 1) {
      int ixj = t ^ j;
      if (ixj > t) {
        float k1 = sk[t], k2 = sk[ixj];
        int   i1 = si[t], i2 = si[ixj];
        bool desc   = ((t & ksz) == 0);
        bool aFirst = (k1 > k2) || (k1 == k2 && i1 < i2);  // stable descending
        if (desc ? !aFirst : aFirst) {
          sk[t] = k2; sk[ixj] = k1;
          si[t] = i2; si[ixj] = i1;
        }
      }
      __syncthreads();
    }
  }
  // gather top-60 rows, pad 385->416 for the conv1 GEMM
  for (int e = t; e < KTOP * PK; e += NPG) {
    int k = e / PK, c = e - k * PK;
    int node = g * NPG + si[k];
    _Float16 v = (c < TLD) ? cat[(size_t)node * TLD + c] : (_Float16)0.f;
    pooled[((size_t)g * KTOP + k) * PK + c] = v;
  }
}

// ---------------- head: relu+maxpool, conv2, lin2 ----------------
__global__ void k_relu_pool(const float* y1, const float* b1, float* y2) {
  int idx = blockIdx.x * 256 + threadIdx.x;        // NB*16*30
  if (idx >= NB * 16 * 30) return;
  int kk = idx % 30, r = idx / 30;
  int o = r & 15, b = r >> 4;
  float bo = b1[o];
  float a0 = fmaxf(y1[((size_t)b * KTOP + 2 * kk) * 16 + o] + bo, 0.f);
  float a1 = fmaxf(y1[((size_t)b * KTOP + 2 * kk + 1) * 16 + o] + bo, 0.f);
  y2[((size_t)b * 16 + o) * 30 + kk] = fmaxf(a0, a1);
}
__global__ void k_conv2(const float* y2, const float* w2, const float* b2,
                        _Float16* y3h) {
  int idx = blockIdx.x * 256 + threadIdx.x;        // NB*32*26
  if (idx >= NB * 32 * 26) return;
  int t = idx % 26, r = idx / 26;
  int o = r & 31, b = r >> 5;
  float s = b2[o];
#pragma unroll
  for (int i = 0; i < 16; ++i) {
    const float* yr = y2 + ((size_t)b * 16 + i) * 30 + t;
    const float* wr = w2 + ((size_t)o * 16 + i) * 5;
#pragma unroll
    for (int j = 0; j < 5; ++j) s += yr[j] * wr[j];
  }
  y3h[(size_t)b * 832 + o * 26 + t] = (_Float16)fmaxf(s, 0.f);
}
__global__ void k_lin2(const float* l1, const float* l1b,
                       const float* w2, const float* b2, float* out) {
  int b = blockIdx.x * 256 + threadIdx.x;
  if (b >= NB) return;
  float s = b2[0];
#pragma unroll 8
  for (int h = 0; h < 128; ++h)
    s += fmaxf(l1[(size_t)b * 128 + h] + l1b[h], 0.f) * w2[h];
  out[b] = s;
}

// ---------------------------------------------------------------------------
extern "C" void kernel_launch(void* const* d_in, const int* in_sizes, int n_in,
                              void* d_out, int out_size, void* d_ws, size_t ws_size,
                              hipStream_t stream) {
  const float* z_table = (const float*)d_in[0];
  const float* Ws[4]   = {(const float*)d_in[1], (const float*)d_in[3],
                          (const float*)d_in[5], (const float*)d_in[7]};
  const float* bs[4]   = {(const float*)d_in[2], (const float*)d_in[4],
                          (const float*)d_in[6], (const float*)d_in[8]};
  const float* conv1_w = (const float*)d_in[9];
  const float* conv1_b = (const float*)d_in[10];
  const float* conv2_w = (const float*)d_in[11];
  const float* conv2_b = (const float*)d_in[12];
  const float* lin1_w  = (const float*)d_in[13];
  const float* lin1_b  = (const float*)d_in[14];
  const float* lin2_w  = (const float*)d_in[15];
  const float* lin2_b  = (const float*)d_in[16];
  const float* ew      = (const float*)d_in[17];
  const int*   z       = (const int*)d_in[18];
  const int*   ei      = (const int*)d_in[19];
  const int*   srcv = ei;
  const int*   dstv = ei + EDG;

  // ---- workspace carve-up (256B aligned) ----
  char* base = (char*)d_ws;
  size_t off = 0;
  auto carve = [&](size_t bytes) -> char* {
    char* p = base + off;
    off = (off + bytes + 255) & ~(size_t)255;
    return p;
  };
  _Float16* xh     = (_Float16*)carve((size_t)NTOT * HID * 2);
  float*    hbuf   = (float*)   carve((size_t)NTOT * HID * 4);
  _Float16* cat    = (_Float16*)carve((size_t)NTOT * TLD * 2);
  float*    key    = (float*)   carve((size_t)NTOT * 4);
  float*    dinv   = (float*)   carve((size_t)NTOT * 4);
  _Float16* wt     = (_Float16*)carve((size_t)832 * 128 * 2);
  _Float16* w1t    = (_Float16*)carve((size_t)16 * PK * 2);
  _Float16* pooled = (_Float16*)carve((size_t)MPOOL * PK * 2);
  float*    y1     = (float*)   carve((size_t)MPOOL * 16 * 4);
  float*    y2     = (float*)   carve((size_t)NB * 16 * 30 * 4);
  _Float16* y3h    = (_Float16*)carve((size_t)NB * 832 * 2);
  float*    l1     = (float*)   carve((size_t)NB * 128 * 4);

  // ---- degree -> dinv (per-graph, LDS atomics) ----
  k_deg_graph<<<NB, NPG, 0, stream>>>(ew, dstv, dinv);

  // ---- embedding ----
  k_embed<<<(NTOT * HID + 255) / 256, 256, 0, stream>>>(z_table, z, xh);

  // ---- 4 GCN layers ----
  const size_t aggLds = (size_t)2 * NPG * HID * sizeof(float);  // 256KB
  for (int L = 0; L < 4; ++L) {
    if (L < 3) {
      k_convert_wt<<<(HID * HID + 255) / 256, 256, 0, stream>>>(Ws[L], wt, HID, HID);
      k_wmma_gemm<8><<<dim3(NTOT / 64, 1), 128, 0, stream>>>(
          xh, wt, hbuf, NTOT, HID, HID);
      k_gcn_agg_fused<<<NB, NPG, aggLds, stream>>>(
          hbuf, ew, srcv, dstv, dinv, bs[L], xh, cat, L * HID);
    } else {
      k_dense_w3<<<(NTOT + 255) / 256, 256, 0, stream>>>(xh, Ws[3], hbuf);
      k_gcn_agg1<<<NB, NPG, 0, stream>>>(hbuf, ew, srcv, dstv, dinv, bs[3], cat, key);
    }
  }

  // ---- sort pool ----
  k_sortpool<<<NB, NPG, 0, stream>>>(key, cat, pooled);

  // ---- conv1 as GEMM [30720 x 416] x [416 x 16] ----
  k_conv1_wt<<<(16 * PK + 255) / 256, 256, 0, stream>>>(conv1_w, w1t);
  k_wmma_gemm<1><<<dim3(MPOOL / 64, 1), 128, 0, stream>>>(
      pooled, w1t, y1, MPOOL, 16, PK);
  k_relu_pool<<<(NB * 16 * 30 + 255) / 256, 256, 0, stream>>>(y1, conv1_b, y2);

  // ---- conv2 ----
  k_conv2<<<(NB * 32 * 26 + 255) / 256, 256, 0, stream>>>(y2, conv2_w, conv2_b, y3h);

  // ---- lin1 as GEMM [512 x 832] x [832 x 128] ----
  k_convert_wt<<<(832 * 128 + 255) / 256, 256, 0, stream>>>(lin1_w, wt, 832, 128);
  k_wmma_gemm<8><<<dim3(NB / 64, 1), 128, 0, stream>>>(y3h, wt, l1, NB, 128, 832);

  // ---- lin2 ----
  k_lin2<<<(NB + 255) / 256, 256, 0, stream>>>(l1, lin1_b, lin2_w, lin2_b, (float*)d_out);
}
